// GRPOTransformer_70403103916703
// MI455X (gfx1250) — matrix operand order repaired and checked by reading, not compiled, not run on hardware
//
#include <hip/hip_runtime.h>

// ---------------------------------------------------------------------------
// GRPO transformer layer for MI455X (gfx1250), wave32 + WMMA bf16.
// 128x128 GEMM tiles: 8 WMMAs per wave per K-step. Async global->LDS staging
// via GLOBAL_LOAD_ASYNC_TO_LDS_B128 (ASYNCcnt) where the toolchain exposes it.
// ---------------------------------------------------------------------------

typedef __attribute__((ext_vector_type(16))) __bf16 v16bf;
typedef __attribute__((ext_vector_type(8)))  float  v8f;
typedef __attribute__((ext_vector_type(4)))  int    v4i;

#define B_    8
#define S_    1024
#define D_    1024
#define HID_  4096
#define E_    8
#define NH_   16
#define HD_   64
#define O_    10
#define NTOK  (B_ * S_)

#if defined(__has_builtin)
#if __has_builtin(__builtin_amdgcn_global_load_async_to_lds_b128)
#define ASYNC_LDS 1
#endif
#endif

typedef __attribute__((address_space(1))) v4i* gptr_v4i;  // global int4*
typedef __attribute__((address_space(3))) v4i* lptr_v4i;  // LDS int4*

// 16-byte global -> LDS copy; async (ASYNCcnt) when available.
__device__ __forceinline__ void cp16_g2l(__bf16* lds, const __bf16* g) {
#if defined(ASYNC_LDS)
  __builtin_amdgcn_global_load_async_to_lds_b128((gptr_v4i)(g), (lptr_v4i)(lds),
                                                 0, 0);
#else
  *(uint4*)lds = *(const uint4*)g;
#endif
}

__device__ __forceinline__ void wait_async0() {
#if defined(ASYNC_LDS)
#if __has_builtin(__builtin_amdgcn_s_wait_asynccnt)
  __builtin_amdgcn_s_wait_asynccnt(0);
#else
  asm volatile("s_wait_asynccnt 0x0" ::: "memory");
#endif
#endif
}

__device__ __forceinline__ v8f wmma_bf16(v16bf a, v16bf b, v8f c) {
  // D = A(16x32) * B(32x16) + C, fp32 accum
  return __builtin_amdgcn_wmma_f32_16x16x32_bf16(false, a, false, b, (short)0, c,
                                                 false, false);
}

__device__ __forceinline__ v8f zero8() {
  v8f z;
#pragma unroll
  for (int i = 0; i < 8; ++i) z[i] = 0.f;
  return z;
}

// A fragment (16x32, MxK): lane l16 = row M; lanes 0-15 hold K 0-15;
// lanes 16-31 hold K 16-31. base points at subtile row 0, k-window 0.
__device__ __forceinline__ v16bf ld_afrag(const __bf16* base, int stride,
                                          int kwin, int l16, int half) {
  const __bf16* p = base + l16 * stride + kwin + (half << 4);
  v16bf a;
#pragma unroll
  for (int i = 0; i < 16; ++i) a[i] = p[i];
  return a;
}

// B fragment (32x16, KxN) where storage is [n][k] row-major (contiguous k):
// lane l16 = column N; lanes 0-15 hold K 0-15, lanes 16-31 hold K 16-31.
__device__ __forceinline__ v16bf ld_bfrag_row(const __bf16* base, int stride,
                                              int kwin, int l16, int half) {
  const __bf16* p = base + l16 * stride + kwin + (half << 4);
  v16bf b;
#pragma unroll
  for (int i = 0; i < 16; ++i) b[i] = p[i];
  return b;
}

// B fragment where storage is [k][n] row-major (k is strided).
__device__ __forceinline__ v16bf ld_bfrag_col(const __bf16* base, int stride,
                                              int kwin, int l16, int half) {
  const __bf16* p = base + (kwin + (half << 4)) * stride + l16;
  v16bf b;
#pragma unroll
  for (int i = 0; i < 16; ++i) b[i] = p[i * stride];
  return b;
}

// ---------------------------------------------------------------------------
// LayerNorm: fp32 [rows, D] -> bf16 [rows, D]
// ---------------------------------------------------------------------------
__global__ __launch_bounds__(256) void ln_kernel(const float* __restrict__ x,
                                                 const float* __restrict__ g,
                                                 const float* __restrict__ be,
                                                 __bf16* __restrict__ out) {
  const int row = blockIdx.x;
  const float4 v = ((const float4*)(x + (size_t)row * D_))[threadIdx.x];
  float s = v.x + v.y + v.z + v.w;
  float q = v.x * v.x + v.y * v.y + v.z * v.z + v.w * v.w;
#pragma unroll
  for (int m = 16; m; m >>= 1) { s += __shfl_xor(s, m); q += __shfl_xor(q, m); }
  __shared__ float sh1[8], sh2[8];
  const int w = threadIdx.x >> 5, lane = threadIdx.x & 31;
  if (lane == 0) { sh1[w] = s; sh2[w] = q; }
  __syncthreads();
  float tot = 0.f, totq = 0.f;
#pragma unroll
  for (int i = 0; i < 8; ++i) { tot += sh1[i]; totq += sh2[i]; }
  const float mean = tot * (1.f / D_);
  const float var = totq * (1.f / D_) - mean * mean;
  const float rstd = rsqrtf(var + 1e-5f);
  const int c = threadIdx.x * 4;
  __bf16* o = out + (size_t)row * D_ + c;
  o[0] = (__bf16)((v.x - mean) * rstd * g[c + 0] + be[c + 0]);
  o[1] = (__bf16)((v.y - mean) * rstd * g[c + 1] + be[c + 1]);
  o[2] = (__bf16)((v.z - mean) * rstd * g[c + 2] + be[c + 2]);
  o[3] = (__bf16)((v.w - mean) * rstd * g[c + 3] + be[c + 3]);
}

// ---------------------------------------------------------------------------
// WMMA GEMM:  C[M,N] = A[M,K](bf16) * W[N,K]^T(fp32) + bias
// Block: 256 thr (8 waves), 128x128 tile, K-step 32.
// Waves: 4(M-sub, 32 rows) x 2(N-sub, 64 cols); each wave = 2x4 accumulators,
// so 8 v_wmma per K-step from 6 fragment loads.
//  MODE 0: bf16 store                       (QKV proj)
//  MODE 1: fp32 store, += residual          (attn out-proj + residual)
//  MODE 2: bf16 store of exact GELU         (expert FC1)
//  MODE 3: fp32 accumulate, * rowscale[m]   (expert FC2, gated accumulate)
// ---------------------------------------------------------------------------
template <int MODE>
__global__ __launch_bounds__(256) void gemm_kernel(
    const __bf16* __restrict__ A, const float* __restrict__ W,
    const float* __restrict__ bias, __bf16* __restrict__ outB,
    float* __restrict__ outF, const float* __restrict__ resid,
    const float* __restrict__ rowscale, int M, int N, int K) {
  __shared__ __align__(16) __bf16 As[128][40];
  __shared__ __align__(16) __bf16 Bs[128][40];
  const int tid = threadIdx.x;
  const int lane = tid & 31, w = tid >> 5;
  const int half = lane >> 4, l16 = lane & 15;
  const int wn = (w & 1) << 6;   // wave col base: 0/64
  const int wm = (w >> 1) << 5;  // wave row base: 0/32/64/96
  const int m0 = blockIdx.y << 7, n0 = blockIdx.x << 7;
  const int srow = tid >> 1, scg = (tid & 1) << 4;  // staging row / k-group
  v8f acc[2][4];
#pragma unroll
  for (int i = 0; i < 2; ++i)
#pragma unroll
    for (int f = 0; f < 4; ++f) acc[i][f] = zero8();
  const __bf16* aptr = A + (size_t)(m0 + srow) * K + scg;
  const float* wptr = W + (size_t)(n0 + srow) * K + scg;
  for (int k0 = 0; k0 < K; k0 += 32) {
    // stage A tile: straight bf16 copy -> async LDS path
    cp16_g2l(&As[srow][scg], aptr + k0);
    cp16_g2l(&As[srow][scg + 8], aptr + k0 + 8);
    // stage B tile: fp32 -> bf16 conversion through VGPRs
    const float4 b0 = *(const float4*)(wptr + k0);
    const float4 b1 = *(const float4*)(wptr + k0 + 4);
    const float4 b2 = *(const float4*)(wptr + k0 + 8);
    const float4 b3 = *(const float4*)(wptr + k0 + 12);
    alignas(16) __bf16 bt[16] = {
        (__bf16)b0.x, (__bf16)b0.y, (__bf16)b0.z, (__bf16)b0.w,
        (__bf16)b1.x, (__bf16)b1.y, (__bf16)b1.z, (__bf16)b1.w,
        (__bf16)b2.x, (__bf16)b2.y, (__bf16)b2.z, (__bf16)b2.w,
        (__bf16)b3.x, (__bf16)b3.y, (__bf16)b3.z, (__bf16)b3.w};
    *(uint4*)&Bs[srow][scg] = *(const uint4*)&bt[0];
    *(uint4*)&Bs[srow][scg + 8] = *(const uint4*)&bt[8];
    if (k0 + 32 < K) {  // global_prefetch_b8 of next tiles
      __builtin_prefetch(aptr + k0 + 32, 0, 0);
      __builtin_prefetch(wptr + k0 + 32, 0, 0);
    }
    wait_async0();
    __syncthreads();
    v16bf af[2], bf[4];
#pragma unroll
    for (int i = 0; i < 2; ++i)
      af[i] = ld_afrag(&As[wm + 16 * i][0], 40, 0, l16, half);
#pragma unroll
    for (int f = 0; f < 4; ++f)
      bf[f] = ld_bfrag_row(&Bs[wn + 16 * f][0], 40, 0, l16, half);
#pragma unroll
    for (int i = 0; i < 2; ++i)
#pragma unroll
      for (int f = 0; f < 4; ++f) acc[i][f] = wmma_bf16(af[i], bf[f], acc[i][f]);
    __syncthreads();
  }
#pragma unroll
  for (int f = 0; f < 4; ++f) {
    const int col = n0 + wn + 16 * f + l16;
    const float bv = bias[col];
#pragma unroll
    for (int i = 0; i < 2; ++i) {
#pragma unroll
      for (int r = 0; r < 8; ++r) {
        const int row = m0 + wm + 16 * i + r + (half << 3);
        const float v = acc[i][f][r] + bv;
        const size_t ix = (size_t)row * N + col;
        if (MODE == 0) {
          outB[ix] = (__bf16)v;
        } else if (MODE == 1) {
          outF[ix] = resid[ix] + v;
        } else if (MODE == 2) {
          outB[ix] = (__bf16)(0.5f * v * (1.f + erff(v * 0.70710678f)));
        } else {
          outF[ix] += rowscale[(size_t)row * E_] * v;
        }
      }
    }
  }
}

// ---------------------------------------------------------------------------
// Flash attention. Grid = B*NH*(S/64). Block = 128 thr (4 waves), each wave
// owns 16 query rows. Online softmax; QK^T and PV via WMMA. hd=64, scale 1/8.
// qkv layout: [n, comp*1024 + h*64 + d], bf16.
// ---------------------------------------------------------------------------
__global__ __launch_bounds__(128) void flash_attn_kernel(
    const __bf16* __restrict__ qkv, __bf16* __restrict__ attn_out) {
  __shared__ __align__(16) __bf16 Qs[64][72];
  __shared__ __align__(16) __bf16 Ks[64][72];
  __shared__ __align__(16) __bf16 Vs[64][72];
  __shared__ __align__(16) __bf16 Pb[4][16][72];
  const int tid = threadIdx.x;
  const int w = tid >> 5, lane = tid & 31;
  const int half = lane >> 4, l16 = lane & 15;
  const int blk = blockIdx.x;
  const int qt = blk & 15;
  const int hh = (blk >> 4) & (NH_ - 1);
  const int b = blk >> 8;
  const int q0 = qt * 64;
  const size_t rs = 3 * D_;
  // Stage Q tile (64 rows x cols 0-31), async copy
#pragma unroll
  for (int i = 0; i < 2; ++i) {
    const int u = tid + 128 * i;
    const int r = u >> 2, cg = u & 3;
    const __bf16* p = qkv + (size_t)(b * S_ + q0 + r) * rs + hh * HD_ + cg * 8;
    cp16_g2l(&Qs[r][cg * 8], p);
  }
  v8f o[4];
  float mrow[8], lrow[8];
#pragma unroll
  for (int f = 0; f < 4; ++f)
#pragma unroll
    for (int i = 0; i < 8; ++i) o[f][i] = 0.f;
#pragma unroll
  for (int r = 0; r < 8; ++r) { mrow[r] = -1e30f; lrow[r] = 0.f; }

  for (int j = 0; j < S_ / 64; ++j) {
    __syncthreads();  // previous PV reads complete before K/V overwrite
#pragma unroll
    for (int i = 0; i < 2; ++i) {
      const int u = tid + 128 * i;
      const int r = u >> 2, cg = u & 3;
      const __bf16* kb =
          qkv + (size_t)(b * S_ + j * 64 + r) * rs + D_ + hh * HD_ + cg * 8;
      cp16_g2l(&Ks[r][cg * 8], kb);
      cp16_g2l(&Vs[r][cg * 8], kb + D_);
    }
    wait_async0();
    __syncthreads();
    // S = (Q K^T) * 1/8 : 16 rows x 64 cols per wave (4 fragments)
    v8f sf[4];
#pragma unroll
    for (int f = 0; f < 4; ++f) {
      v8f s = zero8();
#pragma unroll
      for (int kk = 0; kk < 64; kk += 32) {
        const v16bf a = ld_afrag(&Qs[16 * w][0], 72, kk, l16, half);
        const v16bf bb = ld_bfrag_row(&Ks[16 * f][0], 72, kk, l16, half);
        s = wmma_bf16(a, bb, s);
      }
#pragma unroll
      for (int i = 0; i < 8; ++i) s[i] *= 0.125f;
      sf[f] = s;
    }
    // online softmax over the 64 columns of this tile
    float alpha[8];
#pragma unroll
    for (int r = 0; r < 8; ++r) {
      float mx = sf[0][r];
#pragma unroll
      for (int f = 1; f < 4; ++f) mx = fmaxf(mx, sf[f][r]);
#pragma unroll
      for (int m = 1; m < 16; m <<= 1) mx = fmaxf(mx, __shfl_xor(mx, m));
      const float mn = fmaxf(mrow[r], mx);
      alpha[r] = expf(mrow[r] - mn);
      mrow[r] = mn;
    }
    float rsum[8];
#pragma unroll
    for (int r = 0; r < 8; ++r) rsum[r] = 0.f;
#pragma unroll
    for (int f = 0; f < 4; ++f)
#pragma unroll
      for (int r = 0; r < 8; ++r) {
        const float p = expf(sf[f][r] - mrow[r]);
        sf[f][r] = p;
        rsum[r] += p;
      }
#pragma unroll
    for (int r = 0; r < 8; ++r) {
#pragma unroll
      for (int m = 1; m < 16; m <<= 1) rsum[r] += __shfl_xor(rsum[r], m);
      lrow[r] = lrow[r] * alpha[r] + rsum[r];
    }
#pragma unroll
    for (int f = 0; f < 4; ++f)
#pragma unroll
      for (int r = 0; r < 8; ++r) o[f][r] = o[f][r] * alpha[r];
    // P: C-layout regs -> LDS -> A-layout fragments
#pragma unroll
    for (int f = 0; f < 4; ++f)
#pragma unroll
      for (int r = 0; r < 8; ++r)
        Pb[w][r + (half << 3)][(f << 4) + l16] = (__bf16)sf[f][r];
    __syncthreads();
    // O += P V
#pragma unroll
    for (int f2 = 0; f2 < 4; ++f2)
#pragma unroll
      for (int kk = 0; kk < 64; kk += 32) {
        const v16bf a = ld_afrag(&Pb[w][0][0], 72, kk, l16, half);
        const v16bf bb = ld_bfrag_col(&Vs[0][16 * f2], 72, kk, l16, half);
        o[f2] = wmma_bf16(a, bb, o[f2]);
      }
  }
  // normalize and store bf16 [N, D] at head column block
#pragma unroll
  for (int r = 0; r < 8; ++r) {
    const float inv = 1.f / lrow[r];
    const int grow = b * S_ + q0 + 16 * w + r + (half << 3);
    __bf16* op = attn_out + (size_t)grow * D_ + hh * HD_;
#pragma unroll
    for (int f = 0; f < 4; ++f) op[(f << 4) + l16] = (__bf16)(o[f][r] * inv);
  }
}

// ---------------------------------------------------------------------------
// Gating: per token, 8 waves compute 8 expert logits; thread 0 does top-2
// softmax and writes combined weights [N, E] (zeros elsewhere).
// ---------------------------------------------------------------------------
__global__ __launch_bounds__(256) void gate_kernel(const __bf16* __restrict__ x,
                                                   const float* __restrict__ gw,
                                                   const float* __restrict__ gb,
                                                   float* __restrict__ moew) {
  const int n = blockIdx.x;
  const int w = threadIdx.x >> 5, lane = threadIdx.x & 31;
  const __bf16* xr = x + (size_t)n * D_;
  const float* wr = gw + (size_t)w * D_;
  float s = 0.f;
  for (int i = lane; i < D_; i += 32) s += (float)xr[i] * wr[i];
#pragma unroll
  for (int m = 16; m; m >>= 1) s += __shfl_xor(s, m);
  __shared__ float sg[E_];
  if (lane == 0) sg[w] = s + gb[w];
  __syncthreads();
  if (threadIdx.x == 0) {
    int i0 = 0;
    float b0 = sg[0];
    for (int e = 1; e < E_; ++e)
      if (sg[e] > b0) { b0 = sg[e]; i0 = e; }
    int i1 = -1;
    float b1 = -3.4e38f;
    for (int e = 0; e < E_; ++e)
      if (e != i0 && sg[e] > b1) { b1 = sg[e]; i1 = e; }
    const float e1 = expf(b1 - b0);
    const float inv = 1.f / (1.f + e1);
    float* mo = moew + (size_t)n * E_;
#pragma unroll
    for (int e = 0; e < E_; ++e) mo[e] = 0.f;
    mo[i0] = inv;
    mo[i1] = e1 * inv;
  }
}

// mean over sequence: x2 fp32 [B,S,D] -> pooled [B,D]
__global__ __launch_bounds__(256) void pool_kernel(const float* __restrict__ x2,
                                                   float* __restrict__ pooled) {
  const int t = blockIdx.x * 256 + threadIdx.x;  // B*D
  const int b = t >> 10, d = t & (D_ - 1);
  const float* p = x2 + (size_t)b * S_ * D_ + d;
  float s = 0.f;
  for (int i = 0; i < S_; ++i) s += p[(size_t)i * D_];
  pooled[t] = s * (1.f / S_);
}

// final FC: out[b,o] = pooled[b,:] . fc_w[o,:] + fc_b[o]
__global__ __launch_bounds__(128) void fc_kernel(const float* __restrict__ pooled,
                                                 const float* __restrict__ fw,
                                                 const float* __restrict__ fb,
                                                 float* __restrict__ out) {
  const int bo = blockIdx.x;
  const int b = bo / O_, o = bo % O_;
  const float* p = pooled + (size_t)b * D_;
  const float* wr = fw + (size_t)o * D_;
  float s = 0.f;
  for (int i = threadIdx.x; i < D_; i += 128) s += p[i] * wr[i];
#pragma unroll
  for (int m = 16; m; m >>= 1) s += __shfl_xor(s, m);
  __shared__ float sh[4];
  const int w = threadIdx.x >> 5, lane = threadIdx.x & 31;
  if (lane == 0) sh[w] = s;
  __syncthreads();
  if (threadIdx.x == 0) out[bo] = sh[0] + sh[1] + sh[2] + sh[3] + fb[o];
}

// ---------------------------------------------------------------------------
extern "C" void kernel_launch(void* const* d_in, const int* in_sizes, int n_in,
                              void* d_out, int out_size, void* d_ws,
                              size_t ws_size, hipStream_t stream) {
  (void)in_sizes; (void)n_in; (void)out_size; (void)ws_size;
  const float* x      = (const float*)d_in[0];
  const float* ln1_g  = (const float*)d_in[1];
  const float* ln1_b  = (const float*)d_in[2];
  const float* ln2_g  = (const float*)d_in[3];
  const float* ln2_b  = (const float*)d_in[4];
  const float* qkv_w  = (const float*)d_in[5];
  const float* qkv_b  = (const float*)d_in[6];
  const float* out_w  = (const float*)d_in[7];
  const float* out_b  = (const float*)d_in[8];
  const float* gate_w = (const float*)d_in[9];
  const float* gate_b = (const float*)d_in[10];
  const float* ew1    = (const float*)d_in[11];
  const float* eb1    = (const float*)d_in[12];
  const float* ew2    = (const float*)d_in[13];
  const float* eb2    = (const float*)d_in[14];
  const float* fc_w   = (const float*)d_in[15];
  const float* fc_b   = (const float*)d_in[16];
  float* out = (float*)d_out;

  char* ws = (char*)d_ws;
  size_t off = 0;
  auto alloc = [&](size_t bytes) -> void* {
    void* p = ws + off;
    off += (bytes + 255) & ~(size_t)255;
    return p;
  };
  __bf16* xln1  = (__bf16*)alloc((size_t)NTOK * D_ * 2);
  __bf16* qkv   = (__bf16*)alloc((size_t)NTOK * 3 * D_ * 2);
  __bf16* attn  = (__bf16*)alloc((size_t)NTOK * D_ * 2);
  float*  x1    = (float*)alloc((size_t)NTOK * D_ * 4);
  __bf16* xln2  = (__bf16*)alloc((size_t)NTOK * D_ * 2);
  float*  moew  = (float*)alloc((size_t)NTOK * E_ * 4);
  __bf16* hbuf  = (__bf16*)alloc((size_t)NTOK * HID_ * 2);
  float*  x2    = (float*)alloc((size_t)NTOK * D_ * 4);
  float*  pool  = (float*)alloc((size_t)B_ * D_ * 4);

  // LN1 -> QKV -> flash attention -> out-proj (+residual x)
  ln_kernel<<<NTOK, 256, 0, stream>>>(x, ln1_g, ln1_b, xln1);
  gemm_kernel<0><<<dim3(3 * D_ / 128, NTOK / 128), 256, 0, stream>>>(
      xln1, qkv_w, qkv_b, qkv, nullptr, nullptr, nullptr, NTOK, 3 * D_, D_);
  flash_attn_kernel<<<B_ * NH_ * (S_ / 64), 128, 0, stream>>>(qkv, attn);
  gemm_kernel<1><<<dim3(D_ / 128, NTOK / 128), 256, 0, stream>>>(
      attn, out_w, out_b, nullptr, x1, x, nullptr, NTOK, D_, D_);

  // LN2 -> gate -> dense-expert MoE accumulated onto residual
  ln_kernel<<<NTOK, 256, 0, stream>>>(x1, ln2_g, ln2_b, xln2);
  gate_kernel<<<NTOK, 256, 0, stream>>>(xln2, gate_w, gate_b, moew);
  (void)hipMemcpyAsync(x2, x1, (size_t)NTOK * D_ * 4, hipMemcpyDeviceToDevice,
                       stream);
  for (int e = 0; e < E_; ++e) {
    const float* w1e = ew1 + (size_t)e * HID_ * D_;
    const float* b1e = eb1 + (size_t)e * HID_;
    const float* w2e = ew2 + (size_t)e * D_ * HID_;
    const float* b2e = eb2 + (size_t)e * D_;
    gemm_kernel<2><<<dim3(HID_ / 128, NTOK / 128), 256, 0, stream>>>(
        xln2, w1e, b1e, hbuf, nullptr, nullptr, nullptr, NTOK, HID_, D_);
    gemm_kernel<3><<<dim3(D_ / 128, NTOK / 128), 256, 0, stream>>>(
        hbuf, w2e, b2e, nullptr, x2, nullptr, moew + e, NTOK, D_, HID_);
  }

  // mean over sequence + classifier head
  pool_kernel<<<(B_ * D_) / 256, 256, 0, stream>>>(x2, pool);
  fc_kernel<<<B_ * O_, 128, 0, stream>>>(pool, fc_w, fc_b, out);
}
